// gameGAT_12807592476897
// MI455X (gfx1250) — compile-verified, matrix operand order checked by reading, not run on hardware
//
#include <hip/hip_runtime.h>
#include <hip/hip_bf16.h>

// ---------------------------------------------------------------------------
// MI455X (gfx1250) implementation of the gameGAT reference.
//
// Roofline: ~430 GFLOP of GEMM vs ~1.5 GB of intermediate traffic.
// At 23.3 TB/s HBM the memory time (~65us) is far below matrix-core time,
// so the pipeline is compute-bound -> every linear layer runs on
// v_wmma_f32_16x16x32_f16 (f16 operands, f32 accumulation). Attention over
// N=23 nodes is <3% of FLOPs -> VALU per-batch kernel.
//
// Data movement per WMMA (dominant 256x256 layers, per wave):
//   A fragment: 2x global_load_b128 (reused across 16 N-tile WMMAs)
//   B fragment: 2x ds_load_b128 from the f16 weight image staged in LDS
// ---------------------------------------------------------------------------

typedef __attribute__((ext_vector_type(16))) _Float16 v16h;
typedef __attribute__((ext_vector_type(8)))  float    v8f;

#define B_SZ    16384
#define N_NODES 23
#define C_DIM   256
#define M_ROWS  (B_SZ * N_NODES)   // 376832, divisible by 128

union Frag  { v16h v; uint4 q[2]; unsigned u[8]; _Float16 h[16]; };
union Pack2 { unsigned u; _Float16 h[2]; };

// ISA 7.12.2: 16-bit A-matrix 16x32 per-lane K mapping (2 halfs per VGPR).
// lanes 0-15 hold K = {0..7, 16..23}; lanes 16-31 hold K = {8..15, 24..31}.
// => per lane: two contiguous 8-half (16 B) runs at K = kb and kb+16,
//    kb = (lane&16) ? 8 : 0.
__device__ __forceinline__ int kmap(int v, int lane) {
    int kb = (lane & 16) ? 8 : 0;
    return (v < 4) ? (kb + 2 * v) : (kb + 16 + 2 * (v - 4));
}

// ---------------------------------------------------------------------------
// Generic fused GEMM:  C[M,NOUT] = act(A[M,KT] @ W[NOUT,KT]^T + bias) * scale
//   A:  row-major, f32 (A32=true, K=7 padded layer) or f16 (KT%64==0 layers)
//   W:  row-major f32 in global, converted to f16 LDS in K-chunks of 64
//   C:  row-major, f16 (OUT32=false) or f32
//   ACT: 0 = none, 1 = LeakyReLU(0.2)
// 256 threads = 8 waves; each wave owns one 16-row M tile and accumulates
// every N tile (A-fragment reuse across up to 16 WMMAs per K step).
// Grid: M/128 blocks. M must be a multiple of 128 (376832 and 16384 are).
// NOUT must be a power of two (2/64/128/256 here).
// ---------------------------------------------------------------------------
template <int KT, int NOUT, bool A32, int ACT, bool OUT32>
__global__ __launch_bounds__(256) void gemm_wmma(
    const void* __restrict__ Aop, const float* __restrict__ W,
    const float* __restrict__ bias, void* __restrict__ Cout, float scale)
{
    constexpr int NT  = (NOUT + 15) / 16;       // N tiles (<=16)
    constexpr int KC  = 64;                     // K chunk held in LDS
    constexpr int KCH = (KT + KC - 1) / KC;     // chunks

    __shared__ _Float16 Wl[NT * 16][KC];        // <= 256x64 f16 = 32 KB

    const int tid  = threadIdx.x;
    const int wave = tid >> 5;
    const int lane = tid & 31;
    const int rowbase = (blockIdx.x * 8 + wave) * 16;
    const int arow    = rowbase + (lane & 15);  // A row this lane supplies
    const int kb      = (lane & 16) ? 8 : 0;    // per-lane K base (A and B)

    v8f acc[NT] = {};

    for (int kc = 0; kc < KCH; ++kc) {
        const int k0c = kc * KC;
        __syncthreads();
        // ---- Cooperative W -> LDS (f32 -> f16), zero-padded beyond NOUT/KT.
        if constexpr (KT % 4 == 0) {
            // float4 in, packed uint2 (4 x f16) out
            for (int i = tid; i < NT * 16 * (KC / 4); i += 256) {
                const int n  = i / (KC / 4);
                const int kl = (i % (KC / 4)) * 4;
                float4 w = make_float4(0.f, 0.f, 0.f, 0.f);
                if (n < NOUT) w = *(const float4*)(W + (size_t)n * KT + k0c + kl);
                Pack2 p0, p1;
                p0.h[0] = (_Float16)w.x; p0.h[1] = (_Float16)w.y;
                p1.h[0] = (_Float16)w.z; p1.h[1] = (_Float16)w.w;
                uint2 st; st.x = p0.u; st.y = p1.u;
                *(uint2*)(&Wl[n][kl]) = st;
            }
        } else {  // ragged K (layer 1, KT=7): scalar guarded path
            for (int i = tid; i < NT * 16 * (KC / 2); i += 256) {
                const int n  = i / (KC / 2);
                const int kl = (i % (KC / 2)) * 2;
                const int kg = k0c + kl;
                float w0 = (n < NOUT && kg     < KT) ? W[n * KT + kg]     : 0.f;
                float w1 = (n < NOUT && kg + 1 < KT) ? W[n * KT + kg + 1] : 0.f;
                Pack2 p; p.h[0] = (_Float16)w0; p.h[1] = (_Float16)w1;
                *(unsigned*)(&Wl[n][kl]) = p.u;
            }
        }
        // Prefetch next W chunk into cache (global_prefetch_b8).
        if (kc + 1 < KCH)
            __builtin_prefetch(W + (size_t)(tid & (NOUT - 1)) * KT + (kc + 1) * KC, 0, 1);
        __syncthreads();

#pragma unroll
        for (int ks = 0; ks < KC / 32; ++ks) {
            // --- A fragment (loaded once, reused for all NT WMMAs) ---
            Frag a;
            if constexpr (A32) {
                // f32 source with ragged K: per-element guarded cvt path
#pragma unroll
                for (int v = 0; v < 8; ++v) {
                    const int kg = k0c + ks * 32 + kmap(v, lane);
                    const float* A = (const float*)Aop;
                    float a0 = (kg     < KT) ? A[(size_t)arow * KT + kg]     : 0.f;
                    float a1 = (kg + 1 < KT) ? A[(size_t)arow * KT + kg + 1] : 0.f;
                    Pack2 p; p.h[0] = (_Float16)a0; p.h[1] = (_Float16)a1;
                    a.u[v] = p.u;
                }
            } else {
                // f16 source, KT%64==0: exactly two 16-byte loads per lane
                const _Float16* arp =
                    (const _Float16*)Aop + (size_t)arow * KT + k0c + ks * 32 + kb;
                a.q[0] = *(const uint4*)(arp);        // K = kb + 0..7
                a.q[1] = *(const uint4*)(arp + 16);   // K = kb + 16..23
            }
            // --- all N tiles: B fragment = two 16-byte LDS loads ---
#pragma unroll
            for (int t = 0; t < NT; ++t) {
                const int n = t * 16 + (lane & 15);
                const _Float16* brp = &Wl[n][ks * 32 + kb];
                Frag b;
                b.q[0] = *(const uint4*)(brp);
                b.q[1] = *(const uint4*)(brp + 16);
                acc[t] = __builtin_amdgcn_wmma_f32_16x16x32_f16(
                    false, a.v, false, b.v, (short)0, acc[t], false, false);
            }
        }
    }

    // Epilogue: C/D layout (ISA 7.12.2): vgpr r, lane -> m = r + (lane>=16)*8,
    // n = lane&15 (+16*tile).
    const int mrow = rowbase + ((lane & 16) ? 8 : 0);
#pragma unroll
    for (int t = 0; t < NT; ++t) {
        const int n = t * 16 + (lane & 15);
        if (n < NOUT) {
            const float bv = bias[n];
#pragma unroll
            for (int r = 0; r < 8; ++r) {
                float y = acc[t][r] + bv;
                if (ACT == 1) y = (y >= 0.f) ? y : 0.2f * y;
                y *= scale;
                const size_t off = (size_t)(mrow + r) * NOUT + n;
                if constexpr (OUT32) ((float*)Cout)[off] = y;
                else                 ((_Float16*)Cout)[off] = (_Float16)y;
            }
        }
    }
}

// ---------------------------------------------------------------------------
// agg[b,n,c] = sum_m msg[b,m,c] - msg[b,n,c]
// Thread per (b, c-pair): packed 2 x f16 loads/stores, coalesced in c.
// Grid: B*C/512 blocks of 256.
// ---------------------------------------------------------------------------
__global__ __launch_bounds__(256) void agg_kernel(
    const _Float16* __restrict__ msg, _Float16* __restrict__ agg)
{
    const int idx = blockIdx.x * 256 + threadIdx.x;    // over B * C/2
    const int b  = idx >> 7;
    const int c2 = (idx & 127) * 2;
    float v0[N_NODES], v1[N_NODES];
    float s0 = 0.f, s1 = 0.f;
#pragma unroll
    for (int n = 0; n < N_NODES; ++n) {
        Pack2 p;
        p.u = *(const unsigned*)(msg + ((size_t)b * N_NODES + n) * C_DIM + c2);
        v0[n] = (float)p.h[0]; v1[n] = (float)p.h[1];
        s0 += v0[n]; s1 += v1[n];
    }
#pragma unroll
    for (int n = 0; n < N_NODES; ++n) {
        Pack2 p;
        p.h[0] = (_Float16)(s0 - v0[n]);
        p.h[1] = (_Float16)(s1 - v1[n]);
        *(unsigned*)(agg + ((size_t)b * N_NODES + n) * C_DIM + c2) = p.u;
    }
}

// ---------------------------------------------------------------------------
// Per-batch 8-head attention over N=23 nodes (q pre-scaled by 1/sqrt(32)).
// One workgroup per batch; q/k/v tiles staged in 34.5 KB LDS; thread (n,h)
// computes one query row: 23 dots of length 32, softmax, weighted V sum.
// ---------------------------------------------------------------------------
__global__ __launch_bounds__(256) void attn_kernel(
    const _Float16* __restrict__ qh, const _Float16* __restrict__ kh,
    const _Float16* __restrict__ vh, _Float16* __restrict__ ctx)
{
    __shared__ _Float16 qs[N_NODES][C_DIM];
    __shared__ _Float16 ks[N_NODES][C_DIM];
    __shared__ _Float16 vs[N_NODES][C_DIM];
    const size_t base = (size_t)blockIdx.x * N_NODES * C_DIM;
    {   // vectorized staging: 16 B per thread-iteration per tensor
        uint4* q4 = (uint4*)&qs[0][0];
        uint4* k4 = (uint4*)&ks[0][0];
        uint4* v4 = (uint4*)&vs[0][0];
        const uint4* gq = (const uint4*)(qh + base);
        const uint4* gk = (const uint4*)(kh + base);
        const uint4* gv = (const uint4*)(vh + base);
        for (int i = threadIdx.x; i < N_NODES * C_DIM / 8; i += 256) {
            q4[i] = gq[i]; k4[i] = gk[i]; v4[i] = gv[i];
        }
    }
    __syncthreads();
    const int t = threadIdx.x;
    if (t < N_NODES * 8) {
        const int n  = t / 8;
        const int h  = t % 8;
        const int d0 = h * 32;
        float sc[N_NODES];
        float mx = -1e30f;
        for (int m = 0; m < N_NODES; ++m) {
            float s = 0.f;
#pragma unroll
            for (int d = 0; d < 32; ++d)
                s += (float)qs[n][d0 + d] * (float)ks[m][d0 + d];
            sc[m] = s;
            mx = fmaxf(mx, s);
        }
        float sum = 0.f;
        for (int m = 0; m < N_NODES; ++m) { sc[m] = __expf(sc[m] - mx); sum += sc[m]; }
        const float inv = 1.f / sum;
#pragma unroll
        for (int d = 0; d < 32; ++d) {
            float o = 0.f;
            for (int m = 0; m < N_NODES; ++m) o += sc[m] * (float)vs[m][d0 + d];
            ctx[base + (size_t)n * C_DIM + d0 + d] = (_Float16)(o * inv);
        }
    }
}

// ---------------------------------------------------------------------------
// pooled[b,c] = max_n ctxp[b,n,c]   (2 channels per thread, packed f16)
// ---------------------------------------------------------------------------
__global__ __launch_bounds__(256) void maxpool_kernel(
    const _Float16* __restrict__ ctxp, _Float16* __restrict__ pooled)
{
    const int idx = blockIdx.x * 256 + threadIdx.x;    // over B * C/2
    const int b  = idx >> 7;
    const int c2 = (idx & 127) * 2;
    float m0 = -1e30f, m1 = -1e30f;
#pragma unroll
    for (int n = 0; n < N_NODES; ++n) {
        Pack2 p;
        p.u = *(const unsigned*)(ctxp + ((size_t)b * N_NODES + n) * C_DIM + c2);
        m0 = fmaxf(m0, (float)p.h[0]);
        m1 = fmaxf(m1, (float)p.h[1]);
    }
    Pack2 o; o.h[0] = (_Float16)m0; o.h[1] = (_Float16)m1;
    *(unsigned*)(pooled + (size_t)b * C_DIM + c2) = o.u;
}

// ---------------------------------------------------------------------------
// Launch pipeline.
// Workspace: 4 ping-pong f16 regions of M_ROWS*256 elements (~184 MB each).
// ---------------------------------------------------------------------------
extern "C" void kernel_launch(void* const* d_in, const int* in_sizes, int n_in,
                              void* d_out, int out_size, void* d_ws, size_t ws_size,
                              hipStream_t stream) {
    const float* x      = (const float*)d_in[0];
    const float* msg_w1 = (const float*)d_in[1];
    const float* msg_b1 = (const float*)d_in[2];
    const float* msg_w2 = (const float*)d_in[3];
    const float* msg_b2 = (const float*)d_in[4];
    const float* msg_w3 = (const float*)d_in[5];
    const float* msg_b3 = (const float*)d_in[6];
    const float* q_w    = (const float*)d_in[7];
    const float* q_b    = (const float*)d_in[8];
    const float* k_w    = (const float*)d_in[9];
    const float* k_b    = (const float*)d_in[10];
    const float* v_w    = (const float*)d_in[11];
    const float* v_b    = (const float*)d_in[12];
    const float* in_w   = (const float*)d_in[13];  // [3C, C] packed wq;wk;wv
    const float* in_b   = (const float*)d_in[14];
    const float* out_w  = (const float*)d_in[15];
    const float* out_b  = (const float*)d_in[16];
    const float* o_w1   = (const float*)d_in[17];
    const float* o_b1   = (const float*)d_in[18];
    const float* o_w2   = (const float*)d_in[19];
    const float* o_b2   = (const float*)d_in[20];
    const float* o_w3   = (const float*)d_in[21];
    const float* o_b3   = (const float*)d_in[22];

    char* ws = (char*)d_ws;
    const size_t SZ = (size_t)M_ROWS * C_DIM * sizeof(_Float16);  // ~184 MB
    _Float16* R0 = (_Float16*)(ws);
    _Float16* R1 = (_Float16*)(ws + SZ);
    _Float16* R2 = (_Float16*)(ws + 2 * SZ);
    _Float16* R3 = (_Float16*)(ws + 3 * SZ);

    const dim3 blk(256);
    const dim3 g1(M_ROWS / 128);         // 2944 blocks, 128 rows per block
    const dim3 g2(B_SZ / 128);           // 128 blocks for [B, *] GEMMs
    const dim3 gb(B_SZ);                 // per-batch attention
    const dim3 gc(B_SZ * (C_DIM / 2) / 256);  // packed per-(b,c) kernels
    const float qscale = 0.17677669529663689f;  // 1/sqrt(32)

    // message MLP: 7 -> 128 -> 256 -> 256
    gemm_wmma<  7, 128, true , 1, false><<<g1, blk, 0, stream>>>(x,  msg_w1, msg_b1, R2, 1.f);
    gemm_wmma<128, 256, false, 1, false><<<g1, blk, 0, stream>>>(R2, msg_w2, msg_b2, R1, 1.f);
    gemm_wmma<256, 256, false, 0, false><<<g1, blk, 0, stream>>>(R1, msg_w3, msg_b3, R0, 1.f);
    // sum-aggregate-subtract
    agg_kernel<<<gc, blk, 0, stream>>>(R0, R1);
    // QKV 1x1 convs + LeakyReLU
    gemm_wmma<256, 256, false, 1, false><<<g1, blk, 0, stream>>>(R1, q_w, q_b, R0, 1.f);
    gemm_wmma<256, 256, false, 1, false><<<g1, blk, 0, stream>>>(R1, k_w, k_b, R2, 1.f);
    gemm_wmma<256, 256, false, 1, false><<<g1, blk, 0, stream>>>(R1, v_w, v_b, R3, 1.f);
    // MHA in_proj (q pre-scaled by 1/sqrt(D))
    gemm_wmma<256, 256, false, 0, false><<<g1, blk, 0, stream>>>(R0, in_w,                   in_b,       R1, qscale);
    gemm_wmma<256, 256, false, 0, false><<<g1, blk, 0, stream>>>(R2, in_w + 1 * C_DIM*C_DIM, in_b + 256, R0, 1.f);
    gemm_wmma<256, 256, false, 0, false><<<g1, blk, 0, stream>>>(R3, in_w + 2 * C_DIM*C_DIM, in_b + 512, R2, 1.f);
    // SDPA over N=23 nodes
    attn_kernel<<<gb, blk, 0, stream>>>(R1, R0, R2, R3);
    // out_proj
    gemm_wmma<256, 256, false, 0, false><<<g1, blk, 0, stream>>>(R3, out_w, out_b, R0, 1.f);
    // MaxPool1d(23)
    maxpool_kernel<<<gc, blk, 0, stream>>>(R0, R1);
    // output head: 256 -> 128 -> 64 -> 2 (f32 out)
    gemm_wmma<256, 128, false, 1, false><<<g2, blk, 0, stream>>>(R1, o_w1, o_b1, R2, 1.f);
    gemm_wmma<128,  64, false, 1, false><<<g2, blk, 0, stream>>>(R2, o_w2, o_b2, R3, 1.f);
    gemm_wmma< 64,   2, false, 0, true ><<<g2, blk, 0, stream>>>(R3, o_w3, o_b3, (float*)d_out, 1.f);

    (void)in_sizes; (void)n_in; (void)out_size; (void)ws_size;
}
